// HMLC_39848706572335
// MI455X (gfx1250) — matrix operand order repaired and checked by reading, not compiled, not run on hardware
//
#include <hip/hip_runtime.h>
#include <stdint.h>

typedef __attribute__((ext_vector_type(16))) __bf16 v16bf;
typedef __attribute__((ext_vector_type(8)))  float  v8f;

#define NROW 4096
#define DIM  768
#define BATCH 2048
#define INV_T 14.2857142857142857f

// workspace layout (bytes)
#define OFF_XHI  0
#define OFF_XLO  (NROW*DIM*2)                  // 6,291,456
#define OFF_CODE (OFF_XLO + NROW*DIM*2)        // 12,582,912
#define OFF_STAT (OFF_CODE + 3*BATCH*4)        // +24,576
#define OFF_META (OFF_STAT + 3*3*NROW*4)       // +147,456
// total < 13 MB

// LDS staging buffer layout (per buffer, bytes):
//   Ah[128][40]   at +0      (10,240)
//   Al[128][40]   at +10,240 (10,240)
//   Bh[64][40]    at +20,480 ( 5,120)
//   Bl[64][40]    at +25,600 ( 5,120)
#define BUFSZ   30720
#define SMEMSZ  (2*BUFSZ + 768)   // + ccode[3][64]

__global__ void zero_k(float* p, int n) {
    int i = blockIdx.x * 256 + threadIdx.x;
    if (i < n) p[i] = 0.0f;
}

// features (B,V,D) f32 -> X[(v*B+b), k] as bf16 hi/lo split (round-to-nearest-even)
__global__ void convert_k(const float* __restrict__ f,
                          unsigned short* __restrict__ xhi,
                          unsigned short* __restrict__ xlo) {
    int idx = blockIdx.x * 256 + threadIdx.x;
    if (idx >= NROW * DIM) return;
    int r = idx / DIM, k = idx - r * DIM;
    int b = r & (BATCH - 1), v = r >> 11;
    float x = f[(b * 2 + v) * DIM + k];
    unsigned int u = __float_as_uint(x);
    unsigned int hb = (u + 0x7fffu + ((u >> 16) & 1u)) >> 16;
    float hif = __uint_as_float(hb << 16);
    float resid = x - hif;
    unsigned int u2 = __float_as_uint(resid);
    unsigned int lb = (u2 + 0x7fffu + ((u2 >> 16) & 1u)) >> 16;
    xhi[idx] = (unsigned short)hb;
    xlo[idx] = (unsigned short)lb;
}

// Sequential per-layer mask state: codes[l][b] = valid | nz<<1 | label<<2 ; meta = {nvalid[3], active[3]}
__global__ void masks_k(const int* __restrict__ labels, int* __restrict__ codes,
                        int* __restrict__ meta) {
    __shared__ int svl[BATCH];
    __shared__ unsigned char sval[BATCH], sval2[BATCH];
    __shared__ int s_active, s_nvalid;
    int t = threadIdx.x;
    for (int b = t; b < BATCH; b += 256) sval[b] = 1;
    __syncthreads();
    for (int l = 0; l < 3; ++l) {
        int tc = 2 - l;                       // layer_offset = l+1 -> tc = 2,1,0
        if (t == 0) { s_active = 0; s_nvalid = 0; }
        __syncthreads();
        int la = 0, lv = 0;
        for (int b = t; b < BATCH; b += 256) {
            int v = labels[b * 4 + tc];
            svl[b] = v;
            int vd = sval[b];
            la |= ((v != 0) ? vd : 0);
            lv += vd;
            codes[l * BATCH + b] = vd | ((v != 0) ? 2 : 0) | (v << 2);
        }
        atomicOr(&s_active, la);
        atomicAdd(&s_nvalid, lv);
        __syncthreads();
        if (t == 0) { meta[l] = s_nvalid; meta[3 + l] = s_active; }
        int act = s_active;
        for (int b = t; b < BATCH; b += 256) {
            int vd = sval[b];
            int v = svl[b];
            int nv = vd;
            if (act && vd && v != 0) {
                int first = 1;
                for (int j = 0; j < b; ++j)
                    if (svl[j] == v && sval[j]) { first = 0; break; }
                nv = first;
            }
            sval2[b] = (unsigned char)nv;
        }
        __syncthreads();
        for (int b = t; b < BATCH; b += 256) sval[b] = sval2[b];
        __syncthreads();
    }
}

// async global->LDS 16-byte load (GVS mode: saddr base + 32-bit byte voffset)
__device__ __forceinline__ void async_ld16(unsigned lds_addr,
                                           const unsigned short* gbase,
                                           unsigned voff_bytes) {
    asm volatile("global_load_async_to_lds_b128 %0, %1, %2"
                 :: "v"(lds_addr), "v"(voff_bytes), "s"(gbase)
                 : "memory");
}
__device__ __forceinline__ void wait_async0() {
    asm volatile("s_wait_asynccnt 0x0" ::: "memory");
}

// Fused bf16x3 WMMA GEMM (S = X X^T / T implicit) + per-row/layer {E,P,Sp} stats.
// grid = (32 row-blocks, 8 col-chunks), 256 threads = 8 waves.
// Double-buffered async LDS staging; epilogue S-tile aliases the staging buffers.
__global__ __launch_bounds__(256) void gemm_k(const unsigned short* __restrict__ xhi,
                                              const unsigned short* __restrict__ xlo,
                                              const int* __restrict__ codes,
                                              float* __restrict__ stats) {
    __shared__ __align__(16) unsigned char smem[SMEMSZ];
    float* Sldp  = (float*)smem;                 // [128][65] f32, aliases staging buffers
    int*   ccode = (int*)(smem + 2 * BUFSZ);     // [3][64]

    const int tid = threadIdx.x;
    const int w = tid >> 5, lane = tid & 31;
    const int r0 = blockIdx.x * 128;
    const int cbase = blockIdx.y * 512;

    // staging assignments (6 async b128 loads per thread per K-step)
    const int rowA = tid >> 1;          // 0..127
    const int ccA  = (tid & 1) * 16;    // 0 | 16
    const int rowB = tid >> 2;          // 0..63
    const int ccB  = (tid & 3) * 8;     // 0,8,16,24

    const unsigned smembase = (unsigned)(uintptr_t)(void*)smem;  // LDS byte offset
    const unsigned aLds = smembase + (unsigned)(rowA * 80 + ccA * 2);
    const unsigned bLds = smembase + 20480u + (unsigned)(rowB * 80 + ccB * 2);

    // fragment lane mapping (contiguous k order, same for A and B)
    const int m = lane & 15;
    const int hiHalf = lane >> 4;
    const int ksbA = hiHalf * 16;       // A: lanes<16 -> K 0..15; lanes>=16 -> 16..31 (contiguous)
    const int kbB  = hiHalf * 16;       // B: lanes<16 -> K 0..15; lanes>=16 -> 16..31 (contiguous)

    // stats scan assignment
    const int srow  = tid & 127;
    const int shalf = tid >> 7;
    const int rg = r0 + srow;
    const int bi = rg & (BATCH - 1);
    const int rc0 = codes[0 * BATCH + bi];
    const int rc1 = codes[1 * BATCH + bi];
    const int rc2 = codes[2 * BATCH + bi];

    float e0 = 0, e1 = 0, e2 = 0, p0 = 0, p1 = 0, p2 = 0, q0 = 0, q1 = 0, q2 = 0;

    for (int ci = 0; ci < 8; ++ci) {
        const int c0 = cbase + ci * 64;
        v8f acc[4];
        acc[0] = {}; acc[1] = {}; acc[2] = {}; acc[3] = {};

        // close out previous scan of the aliased S tile before async writes
        __syncthreads();

        // prologue: prefetch K-step 0 into buffer 0
        {
            unsigned aoff = (unsigned)(((r0 + rowA) * DIM + ccA) * 2);
            async_ld16(aLds,           xhi, aoff);
            async_ld16(aLds + 16,      xhi, aoff + 16);
            async_ld16(aLds + 10240,   xlo, aoff);
            async_ld16(aLds + 10256,   xlo, aoff + 16);
            unsigned boff = (unsigned)(((c0 + rowB) * DIM + ccB) * 2);
            async_ld16(bLds,           xhi, boff);
            async_ld16(bLds + 5120,    xlo, boff);
        }

        for (int kc = 0; kc < 24; ++kc) {
            const int cur = kc & 1;
            wait_async0();        // own async writes into buf[cur] complete
            __syncthreads();      // everyone's complete; everyone done reading buf[cur^1]

            if (kc < 23) {        // prefetch next K-step into the other buffer
                const unsigned bo = (unsigned)((cur ^ 1) * BUFSZ);
                const int kg = (kc + 1) * 32;
                unsigned aoff = (unsigned)(((r0 + rowA) * DIM + kg + ccA) * 2);
                async_ld16(aLds + bo,         xhi, aoff);
                async_ld16(aLds + bo + 16,    xhi, aoff + 16);
                async_ld16(aLds + bo + 10240, xlo, aoff);
                async_ld16(aLds + bo + 10256, xlo, aoff + 16);
                unsigned boff = (unsigned)(((c0 + rowB) * DIM + kg + ccB) * 2);
                async_ld16(bLds + bo,         xhi, boff);
                async_ld16(bLds + bo + 5120,  xlo, boff);
            }

            const unsigned short* Ahp = (const unsigned short*)(smem + cur * BUFSZ);
            const unsigned short* Alp = Ahp + 5120;    // +10,240 B
            const unsigned short* Bhp = Ahp + 10240;   // +20,480 B
            const unsigned short* Blp = Ahp + 12800;   // +25,600 B

            union { v16bf v; uint4 q[2]; } aH, aL;
            aH.q[0] = *(const uint4*)(Ahp + (16 * w + m) * 40 + ksbA);
            aH.q[1] = *(const uint4*)(Ahp + (16 * w + m) * 40 + 8 + ksbA);
            aL.q[0] = *(const uint4*)(Alp + (16 * w + m) * 40 + ksbA);
            aL.q[1] = *(const uint4*)(Alp + (16 * w + m) * 40 + 8 + ksbA);

#pragma unroll
            for (int tt = 0; tt < 4; ++tt) {
                union { v16bf v; uint4 q[2]; } bH, bL;
                bH.q[0] = *(const uint4*)(Bhp + (16 * tt + m) * 40 + kbB);
                bH.q[1] = *(const uint4*)(Bhp + (16 * tt + m) * 40 + kbB + 8);
                bL.q[0] = *(const uint4*)(Blp + (16 * tt + m) * 40 + kbB);
                bL.q[1] = *(const uint4*)(Blp + (16 * tt + m) * 40 + kbB + 8);
                // bf16x3: hi*hi + hi*lo + lo*hi (drop lo*lo)
                acc[tt] = __builtin_amdgcn_wmma_f32_16x16x32_bf16(
                    false, aH.v, false, bH.v, (short)0, acc[tt], false, false);
                acc[tt] = __builtin_amdgcn_wmma_f32_16x16x32_bf16(
                    false, aH.v, false, bL.v, (short)0, acc[tt], false, false);
                acc[tt] = __builtin_amdgcn_wmma_f32_16x16x32_bf16(
                    false, aL.v, false, bH.v, (short)0, acc[tt], false, false);
            }
        }

        __syncthreads();   // all frag reads of staging buffers done; S tile may overwrite

        // epilogue: D layout -> LDS tile (row = 16w + d + 8*hiHalf, col = 16t + m)
#pragma unroll
        for (int tt = 0; tt < 4; ++tt)
#pragma unroll
            for (int d = 0; d < 8; ++d)
                Sldp[(16 * w + d + hiHalf * 8) * 65 + 16 * tt + m] = acc[tt][d];

        if (tid < 192) {
            int l = tid >> 6, cc = tid & 63;
            ccode[l * 64 + cc] = codes[l * BATCH + ((c0 + cc) & (BATCH - 1))];
        }
        __syncthreads();

        // scan: thread handles (row srow, 32 cols of its half), 3 layers at once
        for (int c = 0; c < 32; ++c) {
            int cl = shalf * 32 + c;
            int cg = c0 + cl;
            float s = Sldp[srow * 65 + cl] * INV_T;
            float ex = expf(s);
            bool nd = (cg != rg);
            int bc = cg & (BATCH - 1);
            bool eye = (bc == bi);
            {
                int cc = ccode[0 * 64 + cl];
                if (nd && (cc & 1)) {
                    e0 += ex;
                    if (eye || (((cc & rc0 & 2) != 0) && ((cc >> 2) == (rc0 >> 2)))) { p0 += 1.f; q0 += s; }
                }
            }
            {
                int cc = ccode[1 * 64 + cl];
                if (nd && (cc & 1)) {
                    e1 += ex;
                    if (eye || (((cc & rc1 & 2) != 0) && ((cc >> 2) == (rc1 >> 2)))) { p1 += 1.f; q1 += s; }
                }
            }
            {
                int cc = ccode[2 * 64 + cl];
                if (nd && (cc & 1)) {
                    e2 += ex;
                    if (eye || (((cc & rc2 & 2) != 0) && ((cc >> 2) == (rc2 >> 2)))) { p2 += 1.f; q2 += s; }
                }
            }
        }
    }

    atomicAdd(&stats[(0 * 3 + 0) * NROW + rg], e0);
    atomicAdd(&stats[(0 * 3 + 1) * NROW + rg], p0);
    atomicAdd(&stats[(0 * 3 + 2) * NROW + rg], q0);
    atomicAdd(&stats[(1 * 3 + 0) * NROW + rg], e1);
    atomicAdd(&stats[(1 * 3 + 1) * NROW + rg], p1);
    atomicAdd(&stats[(1 * 3 + 2) * NROW + rg], q1);
    atomicAdd(&stats[(2 * 3 + 0) * NROW + rg], e2);
    atomicAdd(&stats[(2 * 3 + 1) * NROW + rg], p2);
    atomicAdd(&stats[(2 * 3 + 2) * NROW + rg], q2);
}

__global__ void final_k(const float* __restrict__ stats, const int* __restrict__ codes,
                        const int* __restrict__ meta, float* __restrict__ out) {
    __shared__ float red[256];
    __shared__ float lloss[3];
    int t = threadIdx.x;
    for (int l = 0; l < 3; ++l) {
        float sum = 0.0f;
        for (int r = t; r < NROW; r += 256) {
            int bi = r & (BATCH - 1);
            if (codes[l * BATCH + bi] & 1) {
                float E  = stats[(l * 3 + 0) * NROW + r];
                float P  = stats[(l * 3 + 1) * NROW + r];
                float Sp = stats[(l * 3 + 2) * NROW + r];
                float ld = (E > 0.0f) ? logf(E) : 0.0f;
                float pp = (P < 1e-6f) ? 1.0f : P;
                sum += -(Sp - P * ld) / pp;   // T/base_T == 1
            }
        }
        red[t] = sum;
        __syncthreads();
        for (int o = 128; o > 0; o >>= 1) {
            if (t < o) red[t] += red[t + o];
            __syncthreads();
        }
        if (t == 0) {
            int nv = meta[l];
            lloss[l] = (nv > 0) ? red[0] / (2.0f * (float)nv) : 0.0f;
        }
        __syncthreads();
    }
    if (t == 0) {
        float cum = 0.0f, nl = 0.0f, maxl = -3.402823466e38f;
        for (int l = 0; l < 3; ++l) {
            float ll = fmaxf(maxl, lloss[l]);
            float pen = exp2f(1.0f / (float)(l + 1));
            if (meta[3 + l]) {
                cum += pen * ll;
                nl += 1.0f;
                maxl = fmaxf(maxl, ll);
            }
        }
        out[0] = cum / nl;
    }
}

extern "C" void kernel_launch(void* const* d_in, const int* in_sizes, int n_in,
                              void* d_out, int out_size, void* d_ws, size_t ws_size,
                              hipStream_t stream) {
    const float* features = (const float*)d_in[0];   // (2048, 2, 768) f32
    const int*   labels   = (const int*)d_in[1];     // (2048, 4) i32
    char* ws = (char*)d_ws;
    unsigned short* xhi = (unsigned short*)(ws + OFF_XHI);
    unsigned short* xlo = (unsigned short*)(ws + OFF_XLO);
    int*   codes = (int*)(ws + OFF_CODE);
    float* stats = (float*)(ws + OFF_STAT);
    int*   meta  = (int*)(ws + OFF_META);

    zero_k<<<(3 * 3 * NROW + 255) / 256, 256, 0, stream>>>(stats, 3 * 3 * NROW);
    convert_k<<<(NROW * DIM + 255) / 256, 256, 0, stream>>>(features, xhi, xlo);
    masks_k<<<1, 256, 0, stream>>>(labels, codes, meta);
    gemm_k<<<dim3(32, 8), 256, 0, stream>>>(xhi, xlo, codes, stats);
    final_k<<<1, 256, 0, stream>>>(stats, codes, meta, (float*)d_out);
}
